// DecoderBlock_16552803959208
// MI455X (gfx1250) — matrix operand order repaired and checked by reading, not compiled, not run on hardware
//
#include <hip/hip_runtime.h>

#define TSEQ  2048
#define NBAT  2
#define CH    1024
#define NHEAD 16
#define HD    64
#define DFFN  4096
#define MROWS (NBAT * TSEQ)   /* 4096 rows = B*T */

typedef __bf16 v16bf __attribute__((ext_vector_type(16)));
typedef float  v8f   __attribute__((ext_vector_type(8)));

union Frag {
  v16bf v;
  uint4 q[2];
  unsigned short u[16];
};

__device__ __forceinline__ unsigned short f2bf(float f) {
  unsigned u = __float_as_uint(f);
  u += 0x7FFFu + ((u >> 16) & 1u);   // round-to-nearest-even
  return (unsigned short)(u >> 16);
}

__device__ __forceinline__ v8f zero8() {
  v8f z;
#pragma unroll
  for (int i = 0; i < 8; ++i) z[i] = 0.0f;
  return z;
}

// xor-swizzle within 16-lane half groups (wave32): and_mask=0x1f, or=0, xor=m
#define SWZF(x, m) __uint_as_float((unsigned)__builtin_amdgcn_ds_swizzle( \
    (int)__float_as_uint(x), (((m) << 10) | 0x1f)))

__device__ __forceinline__ float redmax16(float v) {
  v = fmaxf(v, SWZF(v, 1)); v = fmaxf(v, SWZF(v, 2));
  v = fmaxf(v, SWZF(v, 4)); v = fmaxf(v, SWZF(v, 8));
  return v;
}
__device__ __forceinline__ float redsum16(float v) {
  v += SWZF(v, 1); v += SWZF(v, 2); v += SWZF(v, 4); v += SWZF(v, 8);
  return v;
}

#define WMMA(A, B, C) __builtin_amdgcn_wmma_f32_16x16x32_bf16( \
    false, (A), false, (B), (short)0, (C), false, false)

// ---------------- fp32 -> bf16 (vector x4) ----------------
__global__ __launch_bounds__(256)
void cvt4_kernel(const float* __restrict__ s, unsigned short* __restrict__ d, int n4) {
  int i = blockIdx.x * 256 + threadIdx.x;
  if (i >= n4) return;
  float4 x = ((const float4*)s)[i];
  ushort4 y;
  y.x = f2bf(x.x); y.y = f2bf(x.y); y.z = f2bf(x.z); y.w = f2bf(x.w);
  ((ushort4*)d)[i] = y;
}

// ---------------- bf16 GEMM: out = A[M,K] @ W[K,N] + bias, 128x128x32 tiles
// epilogues: fp32 out / bf16 out / per-head transposed V (bf16 [B,H,64,T])
__global__ __launch_bounds__(256)
void gemm_bf16_kernel(const unsigned short* __restrict__ A,
                      const unsigned short* __restrict__ W,
                      const float* __restrict__ bias,
                      float* __restrict__ outF,
                      unsigned short* __restrict__ outB,
                      unsigned short* __restrict__ outVt,
                      int M, int N, int K, int relu) {
  __shared__ unsigned short As[128][40];   // [m][k], pad 40 avoids bank conflicts
  __shared__ unsigned short Bs[128][40];   // [n][k] (transposed W tile)

  const int nb = N >> 7;
  const int bm = blockIdx.x / nb;
  const int bn = blockIdx.x % nb;
  const int tid  = threadIdx.x;
  const int wave = tid >> 5, lane = tid & 31;
  const int wm = wave >> 2, wn = wave & 3;     // 2 x 4 wave grid, wave = 64M x 32N
  const int hi = lane >> 4, ln = lane & 15;

  v8f acc[4][2];
#pragma unroll
  for (int i = 0; i < 4; ++i)
#pragma unroll
    for (int j = 0; j < 2; ++j) acc[i][j] = zero8();

  const int arow = tid >> 1;           // 0..127
  const int acol = (tid & 1) << 4;     // 0 or 16
  const int brow = tid >> 3;           // 0..31  (k)
  const int bcol = (tid & 7) << 4;     // 0..112 (n)

  const unsigned short* Ag = A + (size_t)(bm * 128 + arow) * K + acol;
  const unsigned short* Wg = W + (size_t)brow * N + bn * 128 + bcol;

  for (int k0 = 0; k0 < K; k0 += 32) {
    uint4 a0 = *(const uint4*)(Ag);
    uint4 a1 = *(const uint4*)(Ag + 8);
    uint4 w0 = *(const uint4*)(Wg);
    uint4 w1 = *(const uint4*)(Wg + 8);
    Ag += 32;
    Wg += (size_t)32 * N;
    __builtin_prefetch(Ag, 0, 1);          // global_prefetch_b8
    __builtin_prefetch(Wg, 0, 1);
    __syncthreads();
    *(uint4*)&As[arow][acol]     = a0;
    *(uint4*)&As[arow][acol + 8] = a1;
    union { uint4 q; unsigned short u[8]; } u0, u1;
    u0.q = w0; u1.q = w1;
#pragma unroll
    for (int i = 0; i < 8; ++i) {
      Bs[bcol + i][brow]     = u0.u[i];
      Bs[bcol + 8 + i][brow] = u1.u[i];
    }
    __syncthreads();

    Frag af[4], bf[2];
#pragma unroll
    for (int im = 0; im < 4; ++im) {
      // A frag (16x32 bf16): lane row = ln, elems = k hi*16 .. hi*16+15
      const unsigned short* r = &As[wm * 64 + im * 16 + ln][hi * 16];
      af[im].q[0] = *(const uint4*)(r);
      af[im].q[1] = *(const uint4*)(r + 8);
    }
#pragma unroll
    for (int in = 0; in < 2; ++in) {
      // B frag (32x16 bf16): lane col = ln, elems = k hi*16 .. hi*16+15
      const unsigned short* r = &Bs[wn * 32 + in * 16 + ln][hi * 16];
      bf[in].q[0] = *(const uint4*)(r);
      bf[in].q[1] = *(const uint4*)(r + 8);
    }
#pragma unroll
    for (int im = 0; im < 4; ++im)
#pragma unroll
      for (int in = 0; in < 2; ++in)
        acc[im][in] = WMMA(af[im].v, bf[in].v, acc[im][in]);
  }

  const size_t rowBase = (size_t)bm * 128 + wm * 64;
  const int colBase = bn * 128 + wn * 32;
#pragma unroll
  for (int im = 0; im < 4; ++im) {
#pragma unroll
    for (int in = 0; in < 2; ++in) {
      const int col = colBase + in * 16 + ln;
      const float bv = bias[col];
#pragma unroll
      for (int r = 0; r < 8; ++r) {
        const size_t row = rowBase + im * 16 + r + hi * 8;  // C layout: M=r(+8 for hi)
        float v = acc[im][in][r] + bv;
        if (relu) v = fmaxf(v, 0.0f);
        const size_t idx = row * (size_t)N + col;
        if (outF) outF[idx] = v;
        if (outB) outB[idx] = f2bf(v);
        if (outVt) {   // V^T: [B, H, 64, T] bf16 for contiguous P@V B-fragments
          const int b = (int)(row >> 11);           // T = 2048
          const int t = (int)(row & (TSEQ - 1));
          const int h = col >> 6, d = col & 63;
          outVt[((size_t)((b * NHEAD + h) * HD + d) << 11) + t] = f2bf(v);
        }
      }
    }
  }
}

// ---------------- causal flash attention: one wave = 16 queries x d=64
__global__ __launch_bounds__(256)
void attn_kernel(const unsigned short* __restrict__ Q,   // [B,T,C] bf16
                 const unsigned short* __restrict__ Kb,  // [B,T,C] bf16
                 const unsigned short* __restrict__ Vt,  // [B,H,64,T] bf16
                 float* __restrict__ O) {                // [B,T,C] fp32
  __shared__ unsigned short P[8][16][40];  // per-wave P tile, 16q x 32k
  const int wave = threadIdx.x >> 5, lane = threadIdx.x & 31;
  const int hi = lane >> 4, ln = lane & 15;
  const int tile = blockIdx.x * 8 + wave;   // 0..4095 = B*H*(T/16)
  const int b   = tile >> 11;               // H*T/16 = 2048
  const int rem = tile & 2047;
  const int h   = rem >> 7;                 // T/16 = 128
  const int q0  = (rem & 127) << 4;

  Frag qf[2];   // Q rows q0..q0+15, d 0-31 and 32-63
  {
    const unsigned short* qp = Q + ((size_t)(b * TSEQ + q0 + ln) * CH + h * HD);
    const int kb16 = hi * 16;
    qf[0].q[0] = *(const uint4*)(qp + kb16);
    qf[0].q[1] = *(const uint4*)(qp + kb16 + 8);
    qf[1].q[0] = *(const uint4*)(qp + 32 + kb16);
    qf[1].q[1] = *(const uint4*)(qp + 32 + kb16 + 8);
  }

  v8f ofr[4];
#pragma unroll
  for (int i = 0; i < 4; ++i) ofr[i] = zero8();
  float mrun[8], lrun[8];
#pragma unroll
  for (int r = 0; r < 8; ++r) { mrun[r] = -__builtin_inff(); lrun[r] = 0.0f; }

  const unsigned short* Kbase = Kb + ((size_t)b * TSEQ) * CH + h * HD;
  const unsigned short* Vbase = Vt + ((size_t)(b * NHEAD + h) * HD) * TSEQ;
  const int kend = q0 + 16;

  for (int kb0 = 0; kb0 < kend; kb0 += 32) {
    v8f s0 = zero8(), s1 = zero8();   // scores: keys kb0.. and kb0+16..
    {
      const unsigned short* kp0 = Kbase + (size_t)(kb0 + ln) * CH + hi * 16;
      const unsigned short* kp1 = kp0 + (size_t)16 * CH;
      Frag bk;
      bk.q[0] = *(const uint4*)(kp0);      bk.q[1] = *(const uint4*)(kp0 + 8);
      s0 = WMMA(qf[0].v, bk.v, s0);
      bk.q[0] = *(const uint4*)(kp0 + 32); bk.q[1] = *(const uint4*)(kp0 + 40);
      s0 = WMMA(qf[1].v, bk.v, s0);
      bk.q[0] = *(const uint4*)(kp1);      bk.q[1] = *(const uint4*)(kp1 + 8);
      s1 = WMMA(qf[0].v, bk.v, s1);
      bk.q[0] = *(const uint4*)(kp1 + 32); bk.q[1] = *(const uint4*)(kp1 + 40);
      s1 = WMMA(qf[1].v, bk.v, s1);
    }

    float alpha[8];
#pragma unroll
    for (int r = 0; r < 8; ++r) {
      const int m = q0 + r + hi * 8;
      float a0 = s0[r] * 0.03125f;          // 1/sqrt(C), C=1024
      float a1 = s1[r] * 0.03125f;
      if (kb0 + ln > m)      a0 = -__builtin_inff();
      if (kb0 + 16 + ln > m) a1 = -__builtin_inff();
      float bmax = redmax16(fmaxf(a0, a1));
      float mnew = fmaxf(mrun[r], bmax);
      float al = __expf(mrun[r] - mnew);
      float p0 = __expf(a0 - mnew);
      float p1 = __expf(a1 - mnew);
      float rs = redsum16(p0 + p1);
      lrun[r] = lrun[r] * al + rs;
      mrun[r] = mnew;
      alpha[r] = al;
      P[wave][r + 8 * hi][ln]      = f2bf(p0);
      P[wave][r + 8 * hi][ln + 16] = f2bf(p1);
    }
#pragma unroll
    for (int d = 0; d < 4; ++d)
#pragma unroll
      for (int r = 0; r < 8; ++r) ofr[d][r] *= alpha[r];

    asm volatile("s_wait_dscnt 0" ::: "memory");  // same-wave LDS RAW fence

    Frag pf;  // re-layout P from C-frag to A-frag via LDS
    pf.q[0] = *(const uint4*)&P[wave][ln][hi * 16];
    pf.q[1] = *(const uint4*)&P[wave][ln][hi * 16 + 8];

#pragma unroll
    for (int db = 0; db < 4; ++db) {
      const unsigned short* vp = Vbase + (size_t)(db * 16 + ln) * TSEQ + kb0 + hi * 16;
      Frag vf;
      vf.q[0] = *(const uint4*)(vp);
      vf.q[1] = *(const uint4*)(vp + 8);
      ofr[db] = WMMA(pf.v, vf.v, ofr[db]);
    }
  }

  float* op = O + ((size_t)(b * TSEQ + q0) * CH + h * HD);
#pragma unroll
  for (int r = 0; r < 8; ++r) {
    const float inv = 1.0f / lrun[r];
    const size_t rowoff = (size_t)(r + 8 * hi) * CH;
#pragma unroll
    for (int db = 0; db < 4; ++db)
      op[rowoff + db * 16 + ln] = ofr[db][r] * inv;
  }
}

// ---------------- out = src + LayerNorm(src)*w ; fp32 and/or bf16 outputs
__global__ __launch_bounds__(256)
void ln_kernel(const float* __restrict__ src, const float* __restrict__ w,
               float* __restrict__ outF, unsigned short* __restrict__ outB) {
  __shared__ float red[256];
  const int row = blockIdx.x;
  const int tid = threadIdx.x;
  const float* sp = src + (size_t)row * CH;
  float4 x = *(const float4*)(sp + tid * 4);

  red[tid] = x.x + x.y + x.z + x.w;
  __syncthreads();
  for (int st = 128; st > 0; st >>= 1) {
    if (tid < st) red[tid] += red[tid + st];
    __syncthreads();
  }
  const float mu = red[0] * (1.0f / CH);
  __syncthreads();

  const float dx0 = x.x - mu, dx1 = x.y - mu, dx2 = x.z - mu, dx3 = x.w - mu;
  red[tid] = dx0 * dx0 + dx1 * dx1 + dx2 * dx2 + dx3 * dx3;
  __syncthreads();
  for (int st = 128; st > 0; st >>= 1) {
    if (tid < st) red[tid] += red[tid + st];
    __syncthreads();
  }
  const float rstd = rsqrtf(red[0] * (1.0f / CH) + 1e-5f);

  const float4 wv = *(const float4*)(w + tid * 4);
  const float o0 = x.x + dx0 * rstd * wv.x;
  const float o1 = x.y + dx1 * rstd * wv.y;
  const float o2 = x.z + dx2 * rstd * wv.z;
  const float o3 = x.w + dx3 * rstd * wv.w;
  if (outF) {
    float4 o; o.x = o0; o.y = o1; o.z = o2; o.w = o3;
    *(float4*)(outF + (size_t)row * CH + tid * 4) = o;
  }
  if (outB) {
    ushort4 o; o.x = f2bf(o0); o.y = f2bf(o1); o.z = f2bf(o2); o.w = f2bf(o3);
    *(ushort4*)(outB + (size_t)row * CH + tid * 4) = o;
  }
}

extern "C" void kernel_launch(void* const* d_in, const int* in_sizes, int n_in,
                              void* d_out, int out_size, void* d_ws, size_t ws_size,
                              hipStream_t stream) {
  (void)in_sizes; (void)n_in; (void)out_size; (void)ws_size;
  const float* x   = (const float*)d_in[0];
  const float* Wq  = (const float*)d_in[1];
  const float* bq  = (const float*)d_in[2];
  const float* Wk  = (const float*)d_in[3];
  const float* bk  = (const float*)d_in[4];
  const float* Wv  = (const float*)d_in[5];
  const float* bv  = (const float*)d_in[6];
  const float* W1  = (const float*)d_in[7];
  const float* b1  = (const float*)d_in[8];
  const float* W2  = (const float*)d_in[9];
  const float* b2  = (const float*)d_in[10];
  const float* lnw = (const float*)d_in[11];

  char* p = (char*)d_ws;
  auto take = [&](size_t bytes) -> char* {
    char* r = p;
    p += (bytes + 255) & ~(size_t)255;
    return r;
  };
  unsigned short* xb  = (unsigned short*)take((size_t)MROWS * CH * 2);
  unsigned short* wqb = (unsigned short*)take((size_t)CH * CH * 2);
  unsigned short* wkb = (unsigned short*)take((size_t)CH * CH * 2);
  unsigned short* wvb = (unsigned short*)take((size_t)CH * CH * 2);
  unsigned short* w1b = (unsigned short*)take((size_t)CH * DFFN * 2);
  unsigned short* w2b = (unsigned short*)take((size_t)DFFN * CH * 2);
  unsigned short* qb  = (unsigned short*)take((size_t)MROWS * CH * 2);
  unsigned short* kb_ = (unsigned short*)take((size_t)MROWS * CH * 2);
  unsigned short* vt  = (unsigned short*)take((size_t)MROWS * CH * 2);
  float*          ab  = (float*)take((size_t)MROWS * CH * 4);
  unsigned short* hb  = (unsigned short*)take((size_t)MROWS * CH * 2);
  unsigned short* g1b = (unsigned short*)take((size_t)MROWS * DFFN * 2);
  float*          fb  = (float*)take((size_t)MROWS * CH * 4);

  auto cvt = [&](const float* s, unsigned short* d, size_t n) {
    int n4 = (int)(n / 4);
    cvt4_kernel<<<(n4 + 255) / 256, 256, 0, stream>>>(s, d, n4);
  };
  auto gemm = [&](const unsigned short* A, const unsigned short* Wb, const float* bias,
                  float* oF, unsigned short* oB, unsigned short* oVt,
                  int M, int N, int K, int relu) {
    gemm_bf16_kernel<<<(M / 128) * (N / 128), 256, 0, stream>>>(
        A, Wb, bias, oF, oB, oVt, M, N, K, relu);
  };

  // 1) cast inputs/weights to bf16
  cvt(x,  xb,  (size_t)MROWS * CH);
  cvt(Wq, wqb, (size_t)CH * CH);
  cvt(Wk, wkb, (size_t)CH * CH);
  cvt(Wv, wvb, (size_t)CH * CH);
  cvt(W1, w1b, (size_t)CH * DFFN);
  cvt(W2, w2b, (size_t)DFFN * CH);

  // 2) QKV projections (bf16 outputs; V written transposed per head)
  gemm(xb, wqb, bq, nullptr, qb, nullptr, MROWS, CH, CH, 0);
  gemm(xb, wkb, bk, nullptr, kb_, nullptr, MROWS, CH, CH, 0);
  gemm(xb, wvb, bv, nullptr, nullptr, vt, MROWS, CH, CH, 0);

  // 3) causal attention -> a (fp32)
  attn_kernel<<<(NBAT * NHEAD * (TSEQ / 16)) / 8, 256, 0, stream>>>(qb, kb_, vt, ab);

  // 4) h = a + LN(a) (bf16 for FFN input)
  ln_kernel<<<MROWS, 256, 0, stream>>>(ab, lnw, nullptr, hb);

  // 5) FFN: relu(h@W1+b1) bf16, then @W2+b2 fp32
  gemm(hb, w1b, b1, nullptr, g1b, nullptr, MROWS, DFFN, CH, 1);
  gemm(g1b, w2b, b2, fb, nullptr, nullptr, MROWS, CH, DFFN, 0);

  // 6) out = f + LN(f)
  ln_kernel<<<MROWS, 256, 0, stream>>>(fb, lnw, (float*)d_out, nullptr);
}
